// QuantumQuanvolutionClassifier_65481071406017
// MI455X (gfx1250) — matrix operand order repaired and checked by reading, not compiled, not run on hardware
//
#include <hip/hip_runtime.h>

// ---------------------------------------------------------------------------
// QuantumQuanvolutionClassifier for MI455X (gfx1250, wave32, WMMA)
//
// Circuit collapses analytically (RZ = phase-only, qubit pairs factorize):
//   e0 = cos(t4)cos(t0)cos(x0) - sin(t4)sin(x0)sin(t1+x1)
//   e1 = cos(t0)cos(x0)cos(t1+x1)
//   e2 = cos(x2)
//   e3 = cos(t3)cos(x2)cos(x3)
// feats[b, 4p+w] = e_w  ->  logits = feats @ W^T + b -> log_softmax
// GEMM via v_wmma_f32_16x16x32_f16 (K padded 784->800, N padded 10->16).
// Features staged in LDS (100KB/block of the 320KB WGP LDS) - no HBM
// round-trip for the intermediate.
// ---------------------------------------------------------------------------

typedef __attribute__((ext_vector_type(16))) _Float16 v16h;
typedef __attribute__((ext_vector_type(8)))  _Float16 v8h;
typedef __attribute__((ext_vector_type(4)))  _Float16 v4h;
typedef __attribute__((ext_vector_type(8)))  float    v8f;

#define KDIM        784
#define KPAD        800      // 25 * 32
#define NPAD        16
#define ROWS_BLK    64       // batch rows per block
#define WAVES_BLK   4        // 4 waves * 16 rows = 64
#define THREADS     128

// ---- pack W [10,784] f32 -> W16 [16][800] f16 (zero padded) ---------------
__global__ void quanv_pack_w(const float* __restrict__ W,
                             _Float16* __restrict__ W16) {
  int idx = blockIdx.x * blockDim.x + threadIdx.x;
  if (idx >= NPAD * KPAD) return;
  int n = idx / KPAD;
  int k = idx - n * KPAD;
  float v = (n < 10 && k < KDIM) ? W[n * KDIM + k] : 0.0f;
  W16[idx] = (_Float16)v;
}

// ---- fused: features (closed form) -> LDS -> WMMA GEMM -> log_softmax -----
extern __shared__ char quanv_smem[];

__global__ __launch_bounds__(THREADS)
void quanv_fused(const float* __restrict__ x,      // [B,28,28]
                 const float* __restrict__ theta,  // [6]
                 const _Float16* __restrict__ W16, // [16][800]
                 const float* __restrict__ bias,   // [10]
                 float* __restrict__ out,          // [B,10]
                 int B) {
  _Float16* feats  = (_Float16*)quanv_smem;                       // [64][800] f16
  float*    logitsT = (float*)(quanv_smem + ROWS_BLK * KPAD * 2); // [4][16][16] f32

  const int tid     = threadIdx.x;
  const int rowBase = blockIdx.x * ROWS_BLK;

  // scalar theta-derived constants (theta[2], theta[5] drop out: RZ is
  // phase-only under Z measurement)
  const float ct0 = cosf(theta[0]);
  const float th1 = theta[1];
  const float ct3 = cosf(theta[3]);
  const float ct4 = cosf(theta[4]);
  const float st4 = sinf(theta[4]);

  // keep the (L2-resident) packed weights warm
  __builtin_prefetch(W16 + (tid << 4), 0, 0);

  // ---- phase 1: closed-form quantum features into LDS ----
  // 64 rows * 200 patch-slots (196 real + 4 zero-pad) = 12800 items
  for (int item = tid; item < ROWS_BLK * 200; item += THREADS) {
    const int r  = item / 200;   // block-relative batch row
    const int ps = item - r * 200;
    _Float16* dst = feats + r * KPAD + ps * 4;
    const int grow = rowBase + r;
    if (ps < 196 && grow < B) {
      const int pr = ps / 14, pc = ps - pr * 14;
      const float* xp = x + (size_t)grow * 784 + (2 * pr) * 28 + 2 * pc;
      const float x0 = xp[0], x1 = xp[1], x2 = xp[28], x3 = xp[29];
      const float cx0 = cosf(x0), sx0 = sinf(x0);
      const float a   = th1 + x1;
      const float ca  = cosf(a),  sa  = sinf(a);
      const float cx2 = cosf(x2), cx3 = cosf(x3);
      const float e0 = ct4 * ct0 * cx0 - st4 * sx0 * sa;
      const float e1 = ct0 * cx0 * ca;
      const float e2 = cx2;
      const float e3 = ct3 * cx2 * cx3;
      v4h v;
      v[0] = (_Float16)e0; v[1] = (_Float16)e1;
      v[2] = (_Float16)e2; v[3] = (_Float16)e3;
      *(v4h*)dst = v;
    } else {
      v4h z = {};
      *(v4h*)dst = z;
    }
  }
  __syncthreads();

  // ---- phase 2: 16x16 tile GEMM per wave, K = 800 = 25 x 32 ----
  const int wave = tid >> 5;
  const int lane = tid & 31;
  const int mn   = lane & 15;                 // M for A, N for B/C/D
  const int aKb  = (lane < 16) ? 0 : 8;       // ISA 7.12.2 16-bit A layout
  const int bKb  = (lane < 16) ? 0 : 8;       // B: same K layout as A
  const _Float16* arow = feats + (wave * 16 + mn) * KPAD;
  const _Float16* brow = W16 + mn * KPAD;

  v8f acc = {};
#pragma unroll 5
  for (int k0 = 0; k0 < KPAD; k0 += 32) {
    const v8h alo = *(const v8h*)(arow + k0 + aKb);        // ds_load_b128
    const v8h ahi = *(const v8h*)(arow + k0 + aKb + 16);
    const v8h blo = *(const v8h*)(brow + k0 + bKb);        // global_load_b128
    const v8h bhi = *(const v8h*)(brow + k0 + bKb + 16);
    v16h A, Bm;
#pragma unroll
    for (int i = 0; i < 8; ++i) {
      A[i]  = alo[i]; A[i + 8]  = ahi[i];
      Bm[i] = blo[i]; Bm[i + 8] = bhi[i];
    }
    acc = __builtin_amdgcn_wmma_f32_16x16x32_f16(
        /*neg_a=*/false, A, /*neg_b=*/false, Bm,
        /*c_mod=*/(short)0, acc, /*reuse_a=*/false, /*reuse_b=*/false);
  }

  // ---- phase 3: scatter C/D fragment (VGPR r -> M = r + 8*(lane>=16)) ----
  float* lt = logitsT + wave * 256;
#pragma unroll
  for (int r = 0; r < 8; ++r) {
    const int mr = r + ((lane < 16) ? 0 : 8);
    lt[mr * 16 + mn] = acc[r];
  }
  __syncthreads();

  // ---- phase 4: bias + log_softmax, one thread per batch row ----
  if (tid < ROWS_BLK && rowBase + tid < B) {
    const float* lrow = logitsT + (tid >> 4) * 256 + (tid & 15) * 16;
    float l[10];
    float mx = -3.0e38f;
#pragma unroll
    for (int n = 0; n < 10; ++n) {
      l[n] = lrow[n] + bias[n];
      mx = fmaxf(mx, l[n]);
    }
    float s = 0.0f;
#pragma unroll
    for (int n = 0; n < 10; ++n) s += expf(l[n] - mx);
    const float lse = mx + logf(s);
    float* o = out + (size_t)(rowBase + tid) * 10;
#pragma unroll
    for (int n = 0; n < 10; ++n) o[n] = l[n] - lse;
  }
}

extern "C" void kernel_launch(void* const* d_in, const int* in_sizes, int n_in,
                              void* d_out, int out_size, void* d_ws, size_t ws_size,
                              hipStream_t stream) {
  (void)n_in; (void)out_size; (void)ws_size;
  const float* x     = (const float*)d_in[0];  // [B,28,28]
  const float* theta = (const float*)d_in[1];  // [6]
  const float* W     = (const float*)d_in[2];  // [10,784]
  const float* bias  = (const float*)d_in[3];  // [10]
  float*       out   = (float*)d_out;          // [B,10]

  const int B = in_sizes[0] / 784;
  _Float16* W16 = (_Float16*)d_ws;             // 16*800*2 = 25.6 KB scratch

  quanv_pack_w<<<(NPAD * KPAD + 255) / 256, 256, 0, stream>>>(W, W16);

  const size_t shmem = (size_t)ROWS_BLK * KPAD * sizeof(_Float16)   // 100 KB feats
                     + (size_t)WAVES_BLK * 16 * 16 * sizeof(float); // 4 KB logits
  const int nBlocks = (B + ROWS_BLK - 1) / ROWS_BLK;
  quanv_fused<<<nBlocks, THREADS, shmem, stream>>>(x, theta, W16, bias, out, B);
}
